// _Expert2_11871289606676
// MI455X (gfx1250) — matrix operand order repaired and checked by reading, not compiled, not run on hardware
//
#include <hip/hip_runtime.h>

#define E_NUM   8
#define TOKENS  4096
#define D_IN    1024
#define D_HID   4096
#define D_OUTV  1024

typedef __attribute__((ext_vector_type(4)))  unsigned v4u;
typedef __attribute__((ext_vector_type(16))) __bf16   v16bf;
typedef __attribute__((ext_vector_type(2)))  __bf16   v2bf;
typedef __attribute__((ext_vector_type(8)))  float    v8f;
typedef __attribute__((ext_vector_type(4)))  float    v4f;
typedef __attribute__((ext_vector_type(2)))  float    v2f;

union FragU { v4u q[2]; v16bf v; };

__device__ __forceinline__ unsigned pack2bf(float lo, float hi) {
  // Lowers to v_cvt_pk_bf16_f32 on gfx1250 (verified in round-3 asm).
  v2f f; f[0] = lo; f[1] = hi;
  union { v2bf v; unsigned u; } c;
  c.v = __builtin_convertvector(f, v2bf);
  return c.u;
}

__device__ __forceinline__ float gelu_tanh(float x) {
  float u = 0.7978845608028654f * (x + 0.044715f * x * x * x);
  return 0.5f * x * (1.0f + tanhf(u));
}

#define LDA 40   // padded ushort stride (rows stay 16B-aligned, avoids bank conflicts)
#define BK  32
#define BN  128

// Elementwise fp32 -> bf16 (RNE), 8 elems/thread, vectorized b128 stores.
__global__ __launch_bounds__(256) void cvt_f32_bf16(
    const float* __restrict__ src, unsigned short* __restrict__ dst, int n)
{
  int i = (blockIdx.x * blockDim.x + threadIdx.x) * 8;
  if (i + 8 > n) return;
  v4f a = *(const v4f*)(src + i);
  v4f b = *(const v4f*)(src + i + 4);
  v4u q;
  q[0] = pack2bf(a[0], a[1]);
  q[1] = pack2bf(a[2], a[3]);
  q[2] = pack2bf(b[0], b[1]);
  q[3] = pack2bf(b[2], b[3]);
  *(v4u*)(dst + i) = q;
}

// Grouped GEMM: Out[m][n] = act( A[m][:] @ W[e][:][n] + bias[e][n] )
// MI: number of 16-row WMMA tiles per wave in M (block M = MI*64).
// A_BF16: A is bf16, else fp32. DO_GELU: fuse gelu, emit bf16.
template<int MI, bool A_BF16, bool DO_GELU>
__global__ __launch_bounds__(256) void ffn_gemm(
    const void* __restrict__ Aptr, const float* __restrict__ W,
    const float* __restrict__ bias, void* __restrict__ Outp,
    const int* __restrict__ cnt, int K, int N)
{
  constexpr int BMT = MI * 64;          // 256 or 128
  constexpr int TPR = 256 / BMT;        // threads per A row: 1 or 2
  constexpr int ACH = BMT / 64;         // v4u chunks per thread for A: 4 or 2

  __shared__ unsigned short As[2][BMT * LDA];
  __shared__ unsigned short Bs[2][BN * LDA];

  const int tid  = threadIdx.x;
  const int lane = tid & 31, wave = tid >> 5;
  const int wx = wave & 1, wy = wave >> 1;      // 4 M-waves x 2 N-waves
  const int hl = lane >> 4, l16 = lane & 15;

  const int m0 = blockIdx.y * BMT;
  const int n0 = blockIdx.x * BN;

  // expert for this M block (counts here are multiples of BMT)
  int e = 0;
  { int start = 0;
    for (int i = 0; i < E_NUM; ++i) { int c = cnt[i];
      if (m0 < start + c) { e = i; break; } start += c; } }

  const float* Wb = W + (size_t)e * K * N;

  // A-tile loader
  const int ar  = tid / TPR;
  const int akb = (tid % TPR) * 16;   // element offset within row (0 when TPR==1)
  // B-tile loader: thread covers 2 K-rows x 8 N-cols (coalesced fp32 reads)
  const int bk = (tid >> 4) * 2;
  const int bn = (tid & 15) * 8;

  v8f acc[MI][4] = {};

  v4f astage_f[2 * ACH]; v4u astage_b[ACH]; v4f bstage[4];
  const int KT = K / BK;

  auto GLOAD = [&](int kt) {
    const int k0 = kt * BK;
    if (A_BF16) {
      const unsigned short* Ab = (const unsigned short*)Aptr + (size_t)(m0 + ar) * K + k0 + akb;
      #pragma unroll
      for (int c = 0; c < ACH; ++c) astage_b[c] = *(const v4u*)(Ab + c * 8);
    } else {
      const float* Af = (const float*)Aptr + (size_t)(m0 + ar) * K + k0 + akb;
      #pragma unroll
      for (int c = 0; c < 2 * ACH; ++c) astage_f[c] = *(const v4f*)(Af + c * 4);
    }
    const float* Wr = Wb + (size_t)(k0 + bk) * N + n0 + bn;
    bstage[0] = *(const v4f*)(Wr);
    bstage[1] = *(const v4f*)(Wr + 4);
    bstage[2] = *(const v4f*)(Wr + N);
    bstage[3] = *(const v4f*)(Wr + N + 4);
  };

  auto LSTORE = [&](int buf) {
    unsigned short* Ad = &As[buf][ar * LDA + akb];
    if (A_BF16) {
      #pragma unroll
      for (int c = 0; c < ACH; ++c) *(v4u*)(Ad + c * 8) = astage_b[c];
    } else {
      #pragma unroll
      for (int c = 0; c < ACH; ++c) {
        v4u q;
        q[0] = pack2bf(astage_f[2 * c][0], astage_f[2 * c][1]);
        q[1] = pack2bf(astage_f[2 * c][2], astage_f[2 * c][3]);
        q[2] = pack2bf(astage_f[2 * c + 1][0], astage_f[2 * c + 1][1]);
        q[3] = pack2bf(astage_f[2 * c + 1][2], astage_f[2 * c + 1][3]);
        *(v4u*)(Ad + c * 8) = q;
      }
    }
    // B stored transposed: Bs[n][k], K packed in pairs per dword
    #pragma unroll
    for (int n = 0; n < 8; ++n) {
      float lo = bstage[n >> 2][n & 3];
      float hi = bstage[2 + (n >> 2)][n & 3];
      *(unsigned*)(&Bs[buf][(bn + n) * LDA + bk]) = pack2bf(lo, hi);
    }
  };

  auto COMPUTE = [&](int buf) {
    FragU fa[MI];
    // A 16-bit 16x32 lane layout: lane half selects K base {0,8}; chunks at +0,+16 elems
    #pragma unroll
    for (int i = 0; i < MI; ++i) {
      const unsigned short* base = &As[buf][(wy * (MI * 16) + i * 16 + l16) * LDA];
      fa[i].q[0] = *(const v4u*)(base + hl * 8);
      fa[i].q[1] = *(const v4u*)(base + hl * 8 + 16);
    }
    // B 32x16 lane layout: lane half selects K base {0,8}; chunks at +0,+16 elems
    #pragma unroll
    for (int j = 0; j < 4; ++j) {
      FragU fb;
      const unsigned short* base = &Bs[buf][(wx * 64 + j * 16 + l16) * LDA];
      fb.q[0] = *(const v4u*)(base + hl * 8);
      fb.q[1] = *(const v4u*)(base + hl * 8 + 16);
      #pragma unroll
      for (int i = 0; i < MI; ++i)
        acc[i][j] = __builtin_amdgcn_wmma_f32_16x16x32_bf16(
            false, fa[i].v, false, fb.v, (short)0, acc[i][j], false, false);
    }
  };

  GLOAD(0); LSTORE(0);
  __syncthreads();
  for (int kt = 0; kt < KT; ++kt) {
    const int cur = kt & 1;
    if (kt + 1 < KT) GLOAD(kt + 1);
    COMPUTE(cur);
    if (kt + 1 < KT) LSTORE(cur ^ 1);
    __syncthreads();
  }

  // Epilogue: C/D layout — element r: M = r + 8*hl, N = l16
  #pragma unroll
  for (int j = 0; j < 4; ++j) {
    const int n = n0 + wx * 64 + j * 16 + l16;
    const float bv = bias[(size_t)e * N + n];
    #pragma unroll
    for (int i = 0; i < MI; ++i) {
      const int mb = m0 + wy * (MI * 16) + i * 16 + hl * 8;
      if (DO_GELU) {
        #pragma unroll
        for (int r = 0; r < 8; r += 2) {
          float v0 = gelu_tanh(acc[i][j][r] + bv);
          float v1 = gelu_tanh(acc[i][j][r + 1] + bv);
          unsigned p = pack2bf(v0, v1);   // one v_cvt_pk_bf16_f32 per pair
          ((unsigned short*)Outp)[(size_t)(mb + r) * N + n]     = (unsigned short)p;
          ((unsigned short*)Outp)[(size_t)(mb + r + 1) * N + n] = (unsigned short)(p >> 16);
        }
      } else {
        #pragma unroll
        for (int r = 0; r < 8; ++r)
          ((float*)Outp)[(size_t)(mb + r) * N + n] = acc[i][j][r] + bv;
      }
    }
  }
}

extern "C" void kernel_launch(void* const* d_in, const int* in_sizes, int n_in,
                              void* d_out, int out_size, void* d_ws, size_t ws_size,
                              hipStream_t stream) {
  const float* inp = (const float*)d_in[0];
  const float* w1  = (const float*)d_in[1];
  const float* b1  = (const float*)d_in[2];
  const float* w2  = (const float*)d_in[3];
  const float* b2  = (const float*)d_in[4];
  const int*   cnt = (const int*)d_in[5];

  unsigned short* h = (unsigned short*)d_ws;         // bf16 [TOKENS][D_HID] = 32 MiB
  float* out = (float*)d_out;

  const size_t need_h = (size_t)TOKENS * D_HID * 2;
  const size_t need_x = (size_t)TOKENS * D_IN * 2;   // 8 MiB

  dim3 blk(256);
  dim3 g1(D_HID / BN, TOKENS / 256);   // 32 x 16 = 512 WGs, BM=256
  dim3 g2(D_OUTV / BN, TOKENS / 128);  //  8 x 32 = 256 WGs, BM=128 (occupancy)

  if (ws_size >= need_h + need_x) {
    // Pre-convert inp to bf16 once (removes 32x-redundant per-N-block conversions)
    unsigned short* xbf = (unsigned short*)((char*)d_ws + need_h);
    const int nx = TOKENS * D_IN;
    cvt_f32_bf16<<<dim3((nx / 8 + 255) / 256), blk, 0, stream>>>(inp, xbf, nx);
    ffn_gemm<4, true, true><<<g1, blk, 0, stream>>>(xbf, w1, b1, h, cnt, D_IN, D_HID);
  } else {
    ffn_gemm<4, false, true><<<g1, blk, 0, stream>>>(inp, w1, b1, h, cnt, D_IN, D_HID);
  }
  ffn_gemm<2, true, false><<<g2, blk, 0, stream>>>(h, w2, b2, out, cnt, D_HID, D_OUTV);
}
